// RoutePredictor_16578573763170
// MI455X (gfx1250) — matrix operand order repaired and checked by reading, not compiled, not run on hardware
//
#include <hip/hip_runtime.h>

// POD vector types (safe in unions, match WMMA builtin signatures)
typedef __attribute__((ext_vector_type(16))) __bf16 v16bf;
typedef __attribute__((ext_vector_type(8)))  float  v8f;
typedef __attribute__((ext_vector_type(4)))  float  f32x4;
typedef __attribute__((ext_vector_type(2)))  float  f32x2;

union V16 { v16bf v; f32x4 f4[2]; };
union V8  { v8f   v; f32x4 f4[2]; };

#define NT 2048
#define NB 512
#define NH 64
#define HSTR 80   // bf16 elements per h row in LDS (160B, 16B-aligned, bank-spread)
#define GSTR 24   // f32 elements per gate row in LDS (96B, 16B-aligned)

// ---- fast activations: native v_tanh_f32 on gfx1250, rcp fallback ----------
#if __has_builtin(__builtin_amdgcn_tanhf)
static __device__ __forceinline__ float fast_tanh(float x) {
  return __builtin_amdgcn_tanhf(x);
}
#else
static __device__ __forceinline__ float fast_tanh(float x) {
  float e = __expf(2.0f * x);                 // v_exp_f32
  return (e - 1.0f) * __builtin_amdgcn_rcpf(e + 1.0f);  // v_rcp_f32, no IEEE divide
}
#endif
static __device__ __forceinline__ float fast_sig(float x) {
  return 0.5f * fast_tanh(0.5f * x) + 0.5f;   // 1 transcendental + 1 fma
}

// B-matrix fragment (32x16 bf16 per K-frag) from f32 weight row-major [256][64].
// W row n is the N-column of B (g = h @ W^T). Holds K [k0..k0+7],[k0+16..k0+23];
// k0 includes +8 for lane>=16 (matches A fragment table order).
static __device__ __forceinline__ v16bf load_bfrag(const float* W, int n, int k0) {
  V16 r;
  const float* p = W + n * NH + k0;
#pragma unroll
  for (int e = 0; e < 16; ++e) r.v[e] = (__bf16)p[e < 8 ? e : e + 8];
  return r.v;
}

// A-matrix fragment (16x32 bf16) from LDS h buffer [16][HSTR] bf16.
// Per ISA table: lane<16 -> K runs [kb..kb+7],[kb+16..kb+23]; lane>=16 -> +8.
static __device__ __forceinline__ v16bf load_afrag(const __bf16* hb, int m, int kb, int lh) {
  V16 r;
  r.f4[0] = *(const f32x4*)(hb + m * HSTR + kb + lh);
  r.f4[1] = *(const f32x4*)(hb + m * HSTR + kb + 16 + lh);
  return r.v;
}

#define WMMA_BF16(A, B, C) \
  __builtin_amdgcn_wmma_f32_16x16x32_bf16(false, (A), false, (B), (short)0, (C), false, false)

__global__ __launch_bounds__(256)
void lstm2_fused(const float* __restrict__ x,
                 const float* __restrict__ w_ih0, const float* __restrict__ w_hh0,
                 const float* __restrict__ b_ih0, const float* __restrict__ b_hh0,
                 const float* __restrict__ w_ih1, const float* __restrict__ w_hh1,
                 const float* __restrict__ b_ih1, const float* __restrict__ b_hh1,
                 const float* __restrict__ fc_w, const float* __restrict__ fc_b,
                 float* __restrict__ out)
{
  __shared__ __align__(16) __bf16 h0buf[16 * HSTR];   // layer0 h state (= layer1 input)
  __shared__ __align__(16) __bf16 h2buf[16 * HSTR];   // layer1 h state (bf16 for WMMA)
  __shared__ __align__(16) float  h2f[16 * NH];       // layer1 h state (f32 for FC)
  __shared__ __align__(16) float  gates[256 * GSTR];  // gate-major [n][m] pre-activations
  __shared__ __align__(16) float  xbuf[16 * 2];       // staged x[t] for this batch tile
  __shared__ __align__(16) float  fcwb[2 * NH];
  __shared__ float fcbb[2];

  const int tid   = threadIdx.x;
  const int lane  = tid & 31;
  const int wv    = tid >> 5;             // wave 0..7
  const int lm    = lane & 15;            // column / row-within-half
  const int lh    = (lane < 16) ? 0 : 8;  // half-select for A / C / D layouts
  const int mbase = blockIdx.x * 16;      // batch tile base

  // ---- one-time init -------------------------------------------------------
  for (int i = tid; i < 16 * HSTR; i += 256) { h0buf[i] = (__bf16)0.0f; h2buf[i] = (__bf16)0.0f; }
  for (int i = tid; i < 2 * NH; i += 256) fcwb[i] = fc_w[i];
  if (tid < 2) fcbb[tid] = fc_b[tid];
  if (tid < 16) {
    const float* xp = x + (size_t)(mbase + tid) * 2;  // t = 0
    xbuf[tid * 2 + 0] = xp[0];
    xbuf[tid * 2 + 1] = xp[1];
  }

  // this wave's two N-tiles (gate columns n0, n1) and K half-offset for B frags
  const int n0 = (2 * wv + 0) * 16 + lm;
  const int n1 = (2 * wv + 1) * 16 + lm;
  const int bk = (lane < 16) ? 0 : 8;

  // persistent weight fragments in VGPRs (bf16): 12 x v16bf = 96 VGPRs
  v16bf whh0_0a = load_bfrag(w_hh0, n0,  0 + bk);
  v16bf whh0_0b = load_bfrag(w_hh0, n0, 32 + bk);
  v16bf whh0_1a = load_bfrag(w_hh0, n1,  0 + bk);
  v16bf whh0_1b = load_bfrag(w_hh0, n1, 32 + bk);
  v16bf wih1_0a = load_bfrag(w_ih1, n0,  0 + bk);
  v16bf wih1_0b = load_bfrag(w_ih1, n0, 32 + bk);
  v16bf wih1_1a = load_bfrag(w_ih1, n1,  0 + bk);
  v16bf wih1_1b = load_bfrag(w_ih1, n1, 32 + bk);
  v16bf whh1_0a = load_bfrag(w_hh1, n0,  0 + bk);
  v16bf whh1_0b = load_bfrag(w_hh1, n0, 32 + bk);
  v16bf whh1_1a = load_bfrag(w_hh1, n1,  0 + bk);
  v16bf whh1_1b = load_bfrag(w_hh1, n1, 32 + bk);

  // layer-0 input projection constants (IN=2) and combined biases, per lane
  const float wx00 = w_ih0[n0 * 2 + 0], wx01 = w_ih0[n0 * 2 + 1];
  const float wx10 = w_ih0[n1 * 2 + 0], wx11 = w_ih0[n1 * 2 + 1];
  const float bb00 = b_ih0[n0] + b_hh0[n0];
  const float bb01 = b_ih0[n1] + b_hh0[n1];
  const float bb10 = b_ih1[n0] + b_hh1[n0];
  const float bb11 = b_ih1[n1] + b_hh1[n1];

  // per-thread c state mapping: fixed hidden unit u, four batch rows m0+4r
  const int u  = tid & 63;
  const int m0 = tid >> 6;
  float c0[4] = {0.f, 0.f, 0.f, 0.f};
  float c1[4] = {0.f, 0.f, 0.f, 0.f};

  __syncthreads();

  for (int t = 0; t < NT; ++t) {
    // ---- layer 0: gates = x@Wih0^T + b + h0@Whh0^T (2 WMMAs per tile) ------
    {
      V8 a0, a1;
#pragma unroll
      for (int j = 0; j < 8; ++j) {                    // D element j -> row m = lh + j
        f32x2 xv = *(const f32x2*)(&xbuf[(lh + j) * 2]);
        a0.v[j] = bb00 + xv[0] * wx00 + xv[1] * wx01;
        a1.v[j] = bb01 + xv[0] * wx10 + xv[1] * wx11;
      }
      v16bf A0 = load_afrag(h0buf, lm, 0, lh);
      v16bf A1 = load_afrag(h0buf, lm, 32, lh);
      a0.v = WMMA_BF16(A0, whh0_0a, a0.v);
      a0.v = WMMA_BF16(A1, whh0_0b, a0.v);
      a1.v = WMMA_BF16(A0, whh0_1a, a1.v);
      a1.v = WMMA_BF16(A1, whh0_1b, a1.v);
      *(f32x4*)(&gates[n0 * GSTR + lh + 0]) = a0.f4[0];
      *(f32x4*)(&gates[n0 * GSTR + lh + 4]) = a0.f4[1];
      *(f32x4*)(&gates[n1 * GSTR + lh + 0]) = a1.f4[0];
      *(f32x4*)(&gates[n1 * GSTR + lh + 4]) = a1.f4[1];
    }
    __syncthreads();

    // ---- layer 0: element-wise LSTM cell ----------------------------------
#pragma unroll
    for (int r = 0; r < 4; ++r) {
      int m = m0 + 4 * r;
      float gi = fast_sig (gates[(0   + u) * GSTR + m]);
      float gf = fast_sig (gates[(64  + u) * GSTR + m]);
      float gg = fast_tanh(gates[(128 + u) * GSTR + m]);
      float go = fast_sig (gates[(192 + u) * GSTR + m]);
      c0[r] = gf * c0[r] + gi * gg;
      float h = go * fast_tanh(c0[r]);
      h0buf[m * HSTR + u] = (__bf16)h;                 // h0_t == layer1 input
    }
    __syncthreads();

    // ---- layer 1: gates = b + h1_t@Wih1^T + h2_{t-1}@Whh1^T (4 WMMAs/tile) -
    {
      V8 a0, a1;
#pragma unroll
      for (int j = 0; j < 8; ++j) { a0.v[j] = bb10; a1.v[j] = bb11; }
      v16bf A0 = load_afrag(h0buf, lm, 0, lh);
      v16bf A1 = load_afrag(h0buf, lm, 32, lh);
      a0.v = WMMA_BF16(A0, wih1_0a, a0.v);
      a0.v = WMMA_BF16(A1, wih1_0b, a0.v);
      a1.v = WMMA_BF16(A0, wih1_1a, a1.v);
      a1.v = WMMA_BF16(A1, wih1_1b, a1.v);
      v16bf B0 = load_afrag(h2buf, lm, 0, lh);
      v16bf B1 = load_afrag(h2buf, lm, 32, lh);
      a0.v = WMMA_BF16(B0, whh1_0a, a0.v);
      a0.v = WMMA_BF16(B1, whh1_0b, a0.v);
      a1.v = WMMA_BF16(B0, whh1_1a, a1.v);
      a1.v = WMMA_BF16(B1, whh1_1b, a1.v);
      *(f32x4*)(&gates[n0 * GSTR + lh + 0]) = a0.f4[0];
      *(f32x4*)(&gates[n0 * GSTR + lh + 4]) = a0.f4[1];
      *(f32x4*)(&gates[n1 * GSTR + lh + 0]) = a1.f4[0];
      *(f32x4*)(&gates[n1 * GSTR + lh + 4]) = a1.f4[1];
    }
    __syncthreads();

    // ---- layer 1: element-wise LSTM cell ----------------------------------
#pragma unroll
    for (int r = 0; r < 4; ++r) {
      int m = m0 + 4 * r;
      float gi = fast_sig (gates[(0   + u) * GSTR + m]);
      float gf = fast_sig (gates[(64  + u) * GSTR + m]);
      float gg = fast_tanh(gates[(128 + u) * GSTR + m]);
      float go = fast_sig (gates[(192 + u) * GSTR + m]);
      c1[r] = gf * c1[r] + gi * gg;
      float h = go * fast_tanh(c1[r]);
      h2buf[m * HSTR + u] = (__bf16)h;                 // for next step's WMMA
      h2f[m * NH + u]     = h;                         // f32 copy for FC
    }
    __syncthreads();

    // ---- FC (OUT=2) + stage x[t+1] ----------------------------------------
    if (tid < 32) {
      int m = tid >> 1, o = tid & 1;
      float s = fcbb[o];
      const f32x4* hv = (const f32x4*)(&h2f[m * NH]);
      const f32x4* wp = (const f32x4*)(&fcwb[o * NH]);
#pragma unroll
      for (int k = 0; k < 16; ++k) {
        f32x4 a = hv[k], b = wp[k];
        s += a[0] * b[0] + a[1] * b[1] + a[2] * b[2] + a[3] * b[3];
      }
      out[((size_t)t * NB + mbase + m) * 2 + o] = s;
    } else if (tid < 48) {
      if (t + 1 < NT) {
        int row = tid - 32;
        const float* xp = x + ((size_t)(t + 1) * NB + mbase + row) * 2;
        xbuf[row * 2 + 0] = xp[0];
        xbuf[row * 2 + 1] = xp[1];
      }
    }
    __syncthreads();
  }
}

extern "C" void kernel_launch(void* const* d_in, const int* in_sizes, int n_in,
                              void* d_out, int out_size, void* d_ws, size_t ws_size,
                              hipStream_t stream) {
  (void)in_sizes; (void)n_in; (void)out_size; (void)d_ws; (void)ws_size;
  lstm2_fused<<<dim3(NB / 16), dim3(256), 0, stream>>>(
      (const float*)d_in[0],                       // x
      (const float*)d_in[1], (const float*)d_in[2],// w_ih0, w_hh0
      (const float*)d_in[3], (const float*)d_in[4],// b_ih0, b_hh0
      (const float*)d_in[5], (const float*)d_in[6],// w_ih1, w_hh1
      (const float*)d_in[7], (const float*)d_in[8],// b_ih1, b_hh1
      (const float*)d_in[9], (const float*)d_in[10],// fc_w, fc_b
      (float*)d_out);
}
